// MultiHeadAttention_88149908783459
// MI455X (gfx1250) — matrix operand order repaired and checked by reading, not compiled, not run on hardware
//
#include <hip/hip_runtime.h>
#include <hip/hip_bf16.h>
#include <math.h>

typedef __attribute__((ext_vector_type(16))) _Float16 v16h;
typedef __attribute__((ext_vector_type(8)))  _Float16 v8h;
typedef __attribute__((ext_vector_type(8)))  float    v8f;
typedef __attribute__((ext_vector_type(4)))  float    v4f;   // native vec for NT stores

#define D_MODEL 1024
#define NH      16
#define DK      64
#define BB      4
#define SS      2048
#define NTOK    (BB*SS)   // 8192
#define SSTR    2052      // padded f32 score row stride (row pitch 8208B, 16B aligned)
#define QSTR    88        // q tile stride (16B-aligned rows, conflict-free ds)

// ---------------- WMMA helpers (CDNA5 gfx1250, wave32) ----------------

static __device__ __forceinline__ v8f wmma16(v16h a, v16h b, v8f c) {
  // D = A(16x32 f16) * B(32x16 f16) + C(16x16 f32)
  return __builtin_amdgcn_wmma_f32_16x16x32_f16(false, a, false, b, (short)0, c,
                                                false, false);
}

// A fragment 16x32 f16: lane<16 -> M=lane, K = {0..15}; lane>=16 -> K+16
static __device__ __forceinline__ v16h a_frag_h(const _Float16* base, int lane, int ld) {
  const int m  = lane & 15;
  const int kb = (lane >> 4) << 4;
  const _Float16* p = base + m * ld + kb;
  v16h a;
#pragma unroll
  for (int j = 0; j < 8; ++j) { a[j] = p[j]; a[8 + j] = p[8 + j]; }
  return a;
}

// Same fragment but converting from an f32 LDS tile (attention probabilities)
static __device__ __forceinline__ v16h a_frag_f(const float* base, int lane, int ld) {
  const int m  = lane & 15;
  const int kb = (lane >> 4) << 4;
  const float* p = base + m * ld + kb;
  v16h a;
#pragma unroll
  for (int j = 0; j < 8; ++j) { a[j] = (_Float16)p[j]; a[8 + j] = (_Float16)p[8 + j]; }
  return a;
}

// B fragment 32x16 f16 from a tile stored [n][k] (i.e. B^T): each lane reads
// 16 contiguous f16 (n = lane&15, k base = (lane>>4)*16)
static __device__ __forceinline__ v16h b_frag_nk(const _Float16* base, int lane, int ld) {
  const int n  = lane & 15;
  const int kb = (lane >> 4) << 4;
  const _Float16* p = base + n * ld + kb;
  v16h b;
#pragma unroll
  for (int j = 0; j < 16; ++j) b[j] = p[j];
  return b;
}

// ---------------- Kernel 1: fused projection  out = (X @ W^T + bias) * scale,
// written head-major f16: out16[b][h][s][d] ----------------

__global__ __launch_bounds__(256) void proj_kernel(
    const float* __restrict__ X, const float* __restrict__ W,
    const float* __restrict__ bias, _Float16* __restrict__ out16, float scale)
{
  __shared__ _Float16 sA[128 * 40];   // X tile  [m][k], padded
  __shared__ _Float16 sB[64 * 40];    // W tile  [n][k], padded

  const int tid  = threadIdx.x;
  const int lane = tid & 31;
  const int wid  = tid >> 5;
  const int wm   = wid >> 1;          // 0..3 : 32-row block
  const int wn   = wid & 1;           // 0..1 : 32-col block
  const int m0   = blockIdx.y * 128;
  const int n0   = blockIdx.x * 64;

  v8f acc[2][2];
#pragma unroll
  for (int i = 0; i < 2; ++i)
#pragma unroll
    for (int j = 0; j < 2; ++j)
#pragma unroll
      for (int v = 0; v < 8; ++v) acc[i][j][v] = 0.f;

  for (int k0 = 0; k0 < 1024; k0 += 32) {
#pragma unroll
    for (int i = 0; i < 4; ++i) {             // A: 128x32 f32, one float4/thread/iter
      int li = tid + 256 * i;
      int r = li >> 3, c4 = li & 7;
      const float4 xv = *(const float4*)&X[(size_t)(m0 + r) * 1024 + k0 + c4 * 4];
      _Float16* p = &sA[r * 40 + c4 * 4];
      p[0] = (_Float16)xv.x; p[1] = (_Float16)xv.y;
      p[2] = (_Float16)xv.z; p[3] = (_Float16)xv.w;
    }
#pragma unroll
    for (int i = 0; i < 2; ++i) {             // B: 64x32 (W rows)
      int li = tid + 256 * i;
      int r = li >> 3, c4 = li & 7;
      const float4 wv = *(const float4*)&W[(size_t)(n0 + r) * 1024 + k0 + c4 * 4];
      _Float16* p = &sB[r * 40 + c4 * 4];
      p[0] = (_Float16)wv.x; p[1] = (_Float16)wv.y;
      p[2] = (_Float16)wv.z; p[3] = (_Float16)wv.w;
    }
    if (k0 + 32 < 1024) {                     // global_prefetch_b8 next tiles
      __builtin_prefetch(&X[(size_t)(m0 + (tid >> 1)) * 1024 + k0 + 32], 0, 1);
      __builtin_prefetch(&W[(size_t)(n0 + (tid & 63)) * 1024 + k0 + 32], 0, 1);
    }
    __syncthreads();
#pragma unroll
    for (int mi = 0; mi < 2; ++mi) {
      v16h af = a_frag_h(sA + (wm * 32 + mi * 16) * 40, lane, 40);
#pragma unroll
      for (int ni = 0; ni < 2; ++ni) {
        v16h bf = b_frag_nk(sB + (wn * 32 + ni * 16) * 40, lane, 40);
        acc[mi][ni] = wmma16(af, bf, acc[mi][ni]);
      }
    }
    __syncthreads();
  }

#pragma unroll
  for (int mi = 0; mi < 2; ++mi) {
#pragma unroll
    for (int ni = 0; ni < 2; ++ni) {
      int ng = n0 + wn * 32 + ni * 16 + (lane & 15);
      int h = ng >> 6, d = ng & 63;
      float bvv = bias[ng];
      int mb = m0 + wm * 32 + mi * 16 + ((lane >> 4) << 3);
#pragma unroll
      for (int v = 0; v < 8; ++v) {
        int mg = mb + v;
        int bi = mg >> 11, s = mg & 2047;
        out16[(((size_t)(bi * NH + h)) * SS + s) * DK + d] =
            (_Float16)((acc[mi][ni][v] + bvv) * scale);
      }
    }
  }
}

// ---------------- Kernel 2: attention for one (b,h,32-query stripe).
// Scores (32x2048 f32) live in CDNA5's 320KB LDS, masked+softmaxed in place,
// attn streamed to d_out (nontemporal), then reused as A for context = P @ V. --

__global__ __launch_bounds__(256) void attn_kernel(
    const _Float16* __restrict__ q16, const _Float16* __restrict__ k16,
    const _Float16* __restrict__ v16, const int* __restrict__ mask,
    float* __restrict__ attn_out, _Float16* __restrict__ ctx16)
{
  __shared__ float    sS[32 * SSTR];     // ~257 KB score stripe
  __shared__ _Float16 sQ[32 * QSTR];     // q tile (pre-scaled by 1/8)
  __shared__ _Float16 sV[64 * 272];      // V^T chunk [d][sk]

  const int tid = threadIdx.x, lane = tid & 31, wid = tid >> 5;
  const int blk = blockIdx.x;
  const int sqt = blk & 63;
  const int h   = (blk >> 6) & 15;
  const int b   = blk >> 10;
  const int bh  = b * NH + h;
  const int sq0 = sqt * 32;

  {                                      // stage 32x64 q tile: one 16B load/thread
    int r = tid >> 3, c8 = tid & 7;
    v8h qv = *(const v8h*)&q16[((size_t)bh * SS + sq0 + r) * DK + c8 * 8];
    *(v8h*)&sQ[r * QSTR + c8 * 8] = qv;
  }
  __syncthreads();

  // -------- scores = q*k^T (scale folded into q) -> LDS --------
  const int wmS = wid >> 2;              // 0..1 : 16-row block
  const int wnS = wid & 3;               // 0..3 : 64-col block
  for (int ch = 0; ch < 8; ++ch) {
    const int n0 = ch * 256 + wnS * 64;
    v8f acc[4];
#pragma unroll
    for (int t = 0; t < 4; ++t)
#pragma unroll
      for (int v = 0; v < 8; ++v) acc[t][v] = 0.f;
#pragma unroll
    for (int ks = 0; ks < 2; ++ks) {
      v16h af = a_frag_h(sQ + (wmS * 16) * QSTR + ks * 32, lane, QSTR);
#pragma unroll
      for (int t = 0; t < 4; ++t) {
        // B[k][n] = k16[key=n][d=k]: 16 contiguous f16, 32B aligned -> b128 loads
        const _Float16* bp = k16 + ((size_t)bh * SS + (n0 + t * 16 + (lane & 15))) * DK
                                 + ks * 32 + ((lane >> 4) << 4);
        v16h bf = *(const v16h*)bp;
        acc[t] = wmma16(af, bf, acc[t]);
      }
    }
#pragma unroll
    for (int t = 0; t < 4; ++t) {
      int nl = n0 + t * 16 + (lane & 15);
      int mb = wmS * 16 + ((lane >> 4) << 3);
#pragma unroll
      for (int v = 0; v < 8; ++v) {
        sS[(mb + v) * SSTR + nl] = acc[t][v];
      }
    }
  }
  __syncthreads();

  // -------- mask + softmax per row (float4 sweeps, wave32 shfl reductions) -----
  for (int r = wid * 4; r < wid * 4 + 4; ++r) {
    float* row = sS + r * SSTR;
    const int4* mrow = (const int4*)(mask + ((size_t)b * SS + sq0 + r) * SS);
    float mx = -3.4e38f;
    for (int c4 = lane; c4 < SS / 4; c4 += 32) {   // mask + running max
      float4 sv = ((float4*)row)[c4];
      int4  mk = mrow[c4];
      sv.x = mk.x ? sv.x : -1.0e9f;
      sv.y = mk.y ? sv.y : -1.0e9f;
      sv.z = mk.z ? sv.z : -1.0e9f;
      sv.w = mk.w ? sv.w : -1.0e9f;
      ((float4*)row)[c4] = sv;
      mx = fmaxf(mx, fmaxf(fmaxf(sv.x, sv.y), fmaxf(sv.z, sv.w)));
    }
#pragma unroll
    for (int off = 16; off; off >>= 1) mx = fmaxf(mx, __shfl_xor(mx, off, 32));
    float sum = 0.f;
    for (int c4 = lane; c4 < SS / 4; c4 += 32) {   // exp + running sum
      float4 sv = ((float4*)row)[c4];
      sv.x = __expf(sv.x - mx); sv.y = __expf(sv.y - mx);
      sv.z = __expf(sv.z - mx); sv.w = __expf(sv.w - mx);
      ((float4*)row)[c4] = sv;
      sum += sv.x + sv.y + sv.z + sv.w;
    }
#pragma unroll
    for (int off = 16; off; off >>= 1) sum += __shfl_xor(sum, off, 32);
    const float inv = 1.0f / sum;
    v4f* out = (v4f*)(attn_out + ((size_t)bh * SS + sq0 + r) * SS);
    for (int c4 = lane; c4 < SS / 4; c4 += 32) {   // normalize + stream out (NT)
      float4 sv = ((float4*)row)[c4];
      sv.x *= inv; sv.y *= inv; sv.z *= inv; sv.w *= inv;
      ((float4*)row)[c4] = sv;
      v4f nv; nv.x = sv.x; nv.y = sv.y; nv.z = sv.z; nv.w = sv.w;
      __builtin_nontemporal_store(nv, out + c4);   // 1GB stream: bypass cache
    }
  }
  __syncthreads();

  // -------- context = P(32x2048) @ V(2048x64), one 16x16 tile per wave --------
  const int wmC = wid >> 2;              // 0..1
  const int wnC = wid & 3;               // 0..3
  v8f cacc;
#pragma unroll
  for (int v = 0; v < 8; ++v) cacc[v] = 0.f;
  for (int ch = 0; ch < 8; ++ch) {
#pragma unroll
    for (int i = 0; i < 8; ++i) {        // stage V^T chunk: 16B loads, transposed stores
      int li = tid + 256 * i;
      int s = li >> 3, d8 = li & 7;
      v8h vv = *(const v8h*)&v16[((size_t)bh * SS + ch * 256 + s) * DK + d8 * 8];
#pragma unroll
      for (int j = 0; j < 8; ++j) sV[(d8 * 8 + j) * 272 + s] = vv[j];
    }
    __syncthreads();
#pragma unroll
    for (int ks = 0; ks < 8; ++ks) {
      v16h af = a_frag_f(sS + (wmC * 16) * SSTR + ch * 256 + ks * 32, lane, SSTR);
      v16h bf = b_frag_nk(sV + (wnC * 16) * 272 + ks * 32, lane, 272);
      cacc = wmma16(af, bf, cacc);
    }
    __syncthreads();
  }
  {
    int d  = wnC * 16 + (lane & 15);
    int mb = wmC * 16 + ((lane >> 4) << 3);
#pragma unroll
    for (int v = 0; v < 8; ++v) {
      int s = sq0 + mb + v;
      ctx16[((size_t)b * SS + s) * D_MODEL + h * DK + d] = (_Float16)cacc[v];
    }
  }
}

// ---------------- Kernel 3: x = ctx @ Wo^T + bo + residual (f32) ----------------

__global__ __launch_bounds__(256) void outproj_kernel(
    const _Float16* __restrict__ A16, const float* __restrict__ W,
    const float* __restrict__ bias, const float* __restrict__ resid,
    float* __restrict__ xout)
{
  __shared__ _Float16 sA[128 * 40];
  __shared__ _Float16 sB[64 * 40];

  const int tid  = threadIdx.x;
  const int lane = tid & 31;
  const int wid  = tid >> 5;
  const int wm   = wid >> 1;
  const int wn   = wid & 1;
  const int m0   = blockIdx.y * 128;
  const int n0   = blockIdx.x * 64;

  v8f acc[2][2];
#pragma unroll
  for (int i = 0; i < 2; ++i)
#pragma unroll
    for (int j = 0; j < 2; ++j)
#pragma unroll
      for (int v = 0; v < 8; ++v) acc[i][j][v] = 0.f;

  for (int k0 = 0; k0 < 1024; k0 += 32) {
#pragma unroll
    for (int i = 0; i < 2; ++i) {             // A: 128x32 f16, one 16B load/thread/iter
      int li = tid + 256 * i;
      int r = li >> 2, c8 = li & 3;
      v8h av = *(const v8h*)&A16[(size_t)(m0 + r) * 1024 + k0 + c8 * 8];
      *(v8h*)&sA[r * 40 + c8 * 8] = av;
    }
#pragma unroll
    for (int i = 0; i < 2; ++i) {             // B: 64x32 f32 (Wo rows)
      int li = tid + 256 * i;
      int r = li >> 3, c4 = li & 7;
      const float4 wv = *(const float4*)&W[(size_t)(n0 + r) * 1024 + k0 + c4 * 4];
      _Float16* p = &sB[r * 40 + c4 * 4];
      p[0] = (_Float16)wv.x; p[1] = (_Float16)wv.y;
      p[2] = (_Float16)wv.z; p[3] = (_Float16)wv.w;
    }
    if (k0 + 32 < 1024) {
      __builtin_prefetch(&A16[(size_t)(m0 + (tid >> 1)) * 1024 + k0 + 32], 0, 1);
      __builtin_prefetch(&W[(size_t)(n0 + (tid & 63)) * 1024 + k0 + 32], 0, 1);
    }
    __syncthreads();
#pragma unroll
    for (int mi = 0; mi < 2; ++mi) {
      v16h af = a_frag_h(sA + (wm * 32 + mi * 16) * 40, lane, 40);
#pragma unroll
      for (int ni = 0; ni < 2; ++ni) {
        v16h bf = b_frag_nk(sB + (wn * 32 + ni * 16) * 40, lane, 40);
        acc[mi][ni] = wmma16(af, bf, acc[mi][ni]);
      }
    }
    __syncthreads();
  }

#pragma unroll
  for (int mi = 0; mi < 2; ++mi) {
#pragma unroll
    for (int ni = 0; ni < 2; ++ni) {
      int ng = n0 + wn * 32 + ni * 16 + (lane & 15);
      float bvv = bias[ng];
      int mb = m0 + wm * 32 + mi * 16 + ((lane >> 4) << 3);
#pragma unroll
      for (int v = 0; v < 8; ++v) {
        int mg = mb + v;
        xout[(size_t)mg * 1024 + ng] =
            acc[mi][ni][v] + bvv + resid[(size_t)mg * 1024 + ng];
      }
    }
  }
}

// ---------------- Kernel 4: layernorm over rows of 1024 ----------------

__global__ __launch_bounds__(256) void ln_kernel(
    const float* __restrict__ x, const float* __restrict__ g,
    const float* __restrict__ be, float* __restrict__ y)
{
  __shared__ float ssum[8], ssq[8];
  const int row = blockIdx.x, tid = threadIdx.x, lane = tid & 31, wid = tid >> 5;
  const float4 xv = ((const float4*)(x + (size_t)row * 1024))[tid];
  float s = xv.x + xv.y + xv.z + xv.w;
  float q = xv.x * xv.x + xv.y * xv.y + xv.z * xv.z + xv.w * xv.w;
#pragma unroll
  for (int off = 16; off; off >>= 1) {
    s += __shfl_xor(s, off, 32);
    q += __shfl_xor(q, off, 32);
  }
  if (lane == 0) { ssum[wid] = s; ssq[wid] = q; }
  __syncthreads();
  float S = 0.f, Q2 = 0.f;
#pragma unroll
  for (int i = 0; i < 8; ++i) { S += ssum[i]; Q2 += ssq[i]; }
  const float mu  = S * (1.0f / 1024.0f);
  const float var = Q2 * (1.0f / 1024.0f) - mu * mu;
  const float inv = rsqrtf(var + 1e-5f);
  const float4 gv = ((const float4*)g)[tid];
  const float4 bv = ((const float4*)be)[tid];
  float4 yv;
  yv.x = (xv.x - mu) * inv * gv.x + bv.x;
  yv.y = (xv.y - mu) * inv * gv.y + bv.y;
  yv.z = (xv.z - mu) * inv * gv.z + bv.z;
  yv.w = (xv.w - mu) * inv * gv.w + bv.w;
  ((float4*)(y + (size_t)row * 1024))[tid] = yv;
}

// ---------------- launch ----------------

extern "C" void kernel_launch(void* const* d_in, const int* in_sizes, int n_in,
                              void* d_out, int out_size, void* d_ws, size_t ws_size,
                              hipStream_t stream) {
  (void)in_sizes; (void)n_in; (void)out_size; (void)ws_size;
  const float* Q  = (const float*)d_in[0];
  const float* K  = (const float*)d_in[1];
  const float* V  = (const float*)d_in[2];
  const int*   M  = (const int*)  d_in[3];
  const float* Wq = (const float*)d_in[4];  const float* bq = (const float*)d_in[5];
  const float* Wk = (const float*)d_in[6];  const float* bk = (const float*)d_in[7];
  const float* Wv = (const float*)d_in[8];  const float* bv = (const float*)d_in[9];
  const float* Wo = (const float*)d_in[10]; const float* bo = (const float*)d_in[11];
  const float* lg = (const float*)d_in[12]; const float* lb = (const float*)d_in[13];

  char* ws = (char*)d_ws;
  _Float16* q16 = (_Float16*)(ws);
  _Float16* k16 = (_Float16*)(ws + (size_t)16 * 1024 * 1024);
  _Float16* v16 = (_Float16*)(ws + (size_t)32 * 1024 * 1024);
  _Float16* c16 = (_Float16*)(ws + (size_t)48 * 1024 * 1024);
  float*    xre = (float*)   (ws + (size_t)64 * 1024 * 1024);

  float* y    = (float*)d_out;                              // (B,S,D)
  float* attn = (float*)d_out + (size_t)NTOK * D_MODEL;     // (B,H,S,S)

  dim3 blk(256);
  dim3 gproj(16, 64);
  hipLaunchKernelGGL(proj_kernel, gproj, blk, 0, stream, Q, Wq, bq, q16, 0.125f);
  hipLaunchKernelGGL(proj_kernel, gproj, blk, 0, stream, K, Wk, bk, k16, 1.0f);
  hipLaunchKernelGGL(proj_kernel, gproj, blk, 0, stream, V, Wv, bv, v16, 1.0f);
  hipLaunchKernelGGL(attn_kernel, dim3(4096), blk, 0, stream, q16, k16, v16, M, attn, c16);
  hipLaunchKernelGGL(outproj_kernel, gproj, blk, 0, stream, c16, Wo, bo, Q, xre);
  hipLaunchKernelGGL(ln_kernel, dim3(8192), blk, 0, stream, xre, lg, lb, y);
}
